// MultiHeadAttention_HJEdit_49555332661325
// MI455X (gfx1250) — matrix operand order repaired and checked by reading, not compiled, not run on hardware
//
#include <hip/hip_runtime.h>
#include <hip/hip_bf16.h>

// Problem constants (from reference)
#define B_  8
#define L_  128
#define T_  64
#define H_  16
#define DH_ 64
#define F_  1024

typedef __attribute__((ext_vector_type(16))) unsigned short v16us;
typedef __attribute__((ext_vector_type(8)))  unsigned short v8us;
typedef __attribute__((ext_vector_type(16))) __bf16         v16bf;
typedef __attribute__((ext_vector_type(8)))  float          v8f;
typedef __attribute__((ext_vector_type(4)))  unsigned int   v4u;
typedef __attribute__((ext_vector_type(8)))  unsigned int   v8u;

static __device__ __forceinline__ unsigned short f2bf(float x) {
  unsigned int u = __builtin_bit_cast(unsigned int, x);
  u += 0x7FFFu + ((u >> 16) & 1u);       // round-to-nearest-even
  return (unsigned short)(u >> 16);
}
static __device__ __forceinline__ v16bf as_bf(v16us u) {
  return __builtin_bit_cast(v16bf, u);
}
static __device__ __forceinline__ v16us cat8(v8us lo, v8us hi) {
  return __builtin_shufflevector(lo, hi, 0,1,2,3,4,5,6,7,8,9,10,11,12,13,14,15);
}
// A fragment (16-bit, 16x32): one contiguous 32B run at k = 2*khalf .. 2*khalf+15
static __device__ __forceinline__ v16us load_a_frag(const unsigned short* row,
                                                    int kk, int khalf) {
  const v8us lo = *(const v8us*)(row + kk + 2 * khalf);
  const v8us hi = *(const v8us*)(row + kk + 2 * khalf + 8);
  return cat8(lo, hi);
}
// B fragment (16-bit, 32x16) from column-major storage: 32B contiguous run
static __device__ __forceinline__ v16us load_b_frag(const unsigned short* colk) {
  const v8us lo = *(const v8us*)(colk);
  const v8us hi = *(const v8us*)(colk + 8);
  return cat8(lo, hi);
}

// ---------------------------------------------------------------------------
// Tensor Data Mover: program a 1-D tile copy (CHUNK_ELEMS f32) global -> LDS.
// D# per CDNA5 ISA: group0 = {count/flags, lds_addr,
// global_addr lo, global_addr hi | type=2}; group1 = {data_size=4B,
// tensor_dim0, tensor_dim1=1, tile_dim0, strides}.  VADDR2/3 = null (2D form).
// ---------------------------------------------------------------------------
#define CH_ 4
#define CHUNK_ELEMS (CH_ * F_)            // 4096 floats = 16 KB
static __device__ __forceinline__ void tdm_load_chunk(unsigned lds_addr,
                                                      unsigned long long gaddr) {
  v4u g0;
  g0[0] = 1u;                                           // count=1 (valid D#)
  g0[1] = lds_addr;                                     // LDS byte address
  g0[2] = (unsigned)(gaddr & 0xFFFFFFFFull);            // global_addr[31:0]
  g0[3] = (unsigned)((gaddr >> 32) & 0x1FFFFFFull)      // global_addr[56:32]
          | (2u << 30);                                 // type = 2 ("image")
  v8u g1;
  g1[0] = 2u << 16;                                     // data_size = 4 bytes
  g1[1] = ((unsigned)CHUNK_ELEMS & 0xFFFFu) << 16;      // tensor_dim0[15:0]
  g1[2] = (((unsigned)CHUNK_ELEMS >> 16) & 0xFFFFu)     // tensor_dim0[31:16]
          | (1u << 16);                                 // tensor_dim1 = 1
  g1[3] = ((unsigned)CHUNK_ELEMS & 0xFFFFu) << 16;      // tile_dim0
  g1[4] = 0u;                                           // tile_dim1/2 unused
  g1[5] = (unsigned)CHUNK_ELEMS;                        // tensor_dim0_stride
  g1[6] = 0u;
  g1[7] = 0u;
  asm volatile("tensor_load_to_lds %0, %1, null, null"
               :: "s"(g0), "s"(g1) : "memory");
}

// ---------------------------------------------------------------------------
// K1: km[b,l,f] = mean_t kv[b,l,t,f]   (pass 1 over kv; store bf16)
//     Double-buffered TDM pipeline: wave 0 programs tensor_load_to_lds for
//     chunk c+1, waits TENSORcnt<=1 for chunk c, whole block reduces from LDS.
// ---------------------------------------------------------------------------
__global__ __launch_bounds__(256) void mean_kv_kernel(
    const float* __restrict__ kv, unsigned short* __restrict__ km) {
  __shared__ float tile[2][CHUNK_ELEMS];   // 2 x 16 KB
  const int bl  = blockIdx.x;              // b*L + l
  const int tid = threadIdx.x;
  const int f4  = tid * 4;
  const float* src = kv + (size_t)bl * T_ * F_;
  const bool issuer = (tid < 32);          // wave 0 drives the TDM
  // LDS byte offsets (flat->LDS truncation; offsets within the group segment)
  const unsigned lds0 = (unsigned)(uintptr_t)(&tile[0][0]);
  const unsigned lds1 = (unsigned)(uintptr_t)(&tile[1][0]);

  if (issuer) {
    tdm_load_chunk(lds0, (unsigned long long)(uintptr_t)src);
  }
  float ax = 0.f, ay = 0.f, az = 0.f, aw = 0.f;
  const int NCH = T_ / CH_;                // 16 chunks
  for (int c = 0; c < NCH; ++c) {
    if (issuer) {
      if (c + 1 < NCH) {
        tdm_load_chunk((c & 1) ? lds0 : lds1,
            (unsigned long long)(uintptr_t)(src + (size_t)(c + 1) * CHUNK_ELEMS));
        __builtin_amdgcn_s_wait_tensorcnt((short)1);   // chunk c complete
      } else {
        __builtin_amdgcn_s_wait_tensorcnt((short)0);
      }
    }
    __syncthreads();
    const float* buf = &tile[c & 1][0];
#pragma unroll
    for (int r = 0; r < CH_; ++r) {
      const float4 v = *(const float4*)(&buf[r * F_ + f4]);
      ax += v.x; ay += v.y; az += v.z; aw += v.w;
    }
    __syncthreads();                       // readers done before overwrite
  }
  const float inv = 1.0f / (float)T_;
  unsigned short* dst = km + (size_t)bl * F_ + f4;
  dst[0] = f2bf(ax * inv); dst[1] = f2bf(ay * inv);
  dst[2] = f2bf(az * inv); dst[3] = f2bf(aw * inv);
}

// ---------------------------------------------------------------------------
// K2: qproj[b,h,d] = Wq[d,:] . q[b, h*64:...] + bq[d]
// ---------------------------------------------------------------------------
__global__ __launch_bounds__(256) void qproj_kernel(
    const float* __restrict__ q, const float* __restrict__ Wq,
    const float* __restrict__ bq, float* __restrict__ qp) {
  const int o = blockIdx.x * 256 + threadIdx.x;   // (b*H + h)*64 + d
  const int d = o & 63;
  const int bh = o >> 6;
  const int h = bh & (H_ - 1);
  const int b = bh >> 4;
  const float* qrow = q  + (size_t)b * F_ + h * DH_;
  const float* wrow = Wq + (size_t)d * DH_;
  float acc = bq[d];
  for (int k = 0; k < DH_; ++k) acc += wrow[k] * qrow[k];
  qp[o] = acc;
}

// ---------------------------------------------------------------------------
// K3: Wk -> bf16, natural [d][k] layout (== column-major B for kproj GEMM)
// ---------------------------------------------------------------------------
__global__ __launch_bounds__(256) void wkpack_kernel(
    const float* __restrict__ Wk, unsigned short* __restrict__ WkBt) {
  const int o = blockIdx.x * 256 + threadIdx.x;   // d*64 + k
  WkBt[o] = f2bf(Wk[o]);
}

// ---------------------------------------------------------------------------
// K4: fold Wv through softmax into Wo:
//     W2[i, h*64+k] = sum_d Wo[i, h*64+d] * Wv[d, k]  (natural [i][k] bf16)
//     bvec[i]       = bo[i] + sum_{h,d} Wo[i, h*64+d] * bv[d]
// ---------------------------------------------------------------------------
__global__ __launch_bounds__(64) void wcomb_kernel(
    const float* __restrict__ Wv, const float* __restrict__ bv,
    const float* __restrict__ Wo, const float* __restrict__ bo,
    unsigned short* __restrict__ W2bf, float* __restrict__ bvec) {
  const int i = blockIdx.x;
  const int t = threadIdx.x;                      // k index
  const float* worow = Wo + (size_t)i * F_;
  float bacc = 0.f;
  for (int h = 0; h < H_; ++h) {
    const float* wseg = worow + h * DH_;
    float acc = 0.f;
    for (int d = 0; d < DH_; ++d) acc += wseg[d] * Wv[(size_t)d * DH_ + t];
    W2bf[(size_t)i * F_ + h * DH_ + t] = f2bf(acc);
    bacc += wseg[t] * bv[t];
  }
  __shared__ float red[64];
  red[t] = bacc;
  __syncthreads();
  for (int s = 32; s > 0; s >>= 1) {
    if (t < s) red[t] += red[t + s];
    __syncthreads();
  }
  if (t == 0) bvec[i] = bo[i] + red[0];
}

// ---------------------------------------------------------------------------
// K5: K-projection GEMM via WMMA bf16 (M=16384, K=64, N=64)
// ---------------------------------------------------------------------------
__global__ __launch_bounds__(128) void kproj_wmma_kernel(
    const unsigned short* __restrict__ km, const unsigned short* __restrict__ WkBt,
    const float* __restrict__ bk, float* __restrict__ kh) {
  const int lane  = threadIdx.x & 31;
  const int wave  = threadIdx.x >> 5;
  const int mtile = blockIdx.x * 4 + wave;
  const int m0    = mtile * 16;
  const int khalf = (lane >= 16) ? 8 : 0;
  const int kb0   = (lane >= 16) ? 16 : 0;

  const unsigned short* arow = km + (size_t)(m0 + (lane & 15)) * DH_;
  const v16bf a0 = as_bf(load_a_frag(arow, 0, khalf));
  const v16bf a1 = as_bf(load_a_frag(arow, 32, khalf));

#pragma unroll
  for (int ns = 0; ns < 4; ++ns) {
    const int ncol = ns * 16 + (lane & 15);
    const unsigned short* bcol = WkBt + (size_t)ncol * DH_;
    v8f c = (v8f)0.0f;
    c = __builtin_amdgcn_wmma_f32_16x16x32_bf16(false, a0, false,
            as_bf(load_b_frag(bcol + kb0)), (short)0, c, false, false);
    c = __builtin_amdgcn_wmma_f32_16x16x32_bf16(false, a1, false,
            as_bf(load_b_frag(bcol + 32 + kb0)), (short)0, c, false, false);
#pragma unroll
    for (int r = 0; r < 8; ++r) {
      const int m = m0 + r + ((lane >= 16) ? 8 : 0);
      const int n = ns * 16 + (lane & 15);
      kh[(size_t)m * DH_ + n] = c[r] + bk[n];
    }
  }
}

// ---------------------------------------------------------------------------
// K6: logits + softmax. One block per (b1,h,b2), 128 threads = L.
// ---------------------------------------------------------------------------
__global__ __launch_bounds__(128) void attn_softmax_kernel(
    const float* __restrict__ qp, const float* __restrict__ kh,
    unsigned short* __restrict__ attn) {
  const int blk = blockIdx.x;                      // (b1*H + h)*B + b2
  const int b2 = blk & (B_ - 1);
  const int h  = (blk >> 3) & (H_ - 1);
  const int b1 = blk >> 7;
  const int l  = threadIdx.x;
  const float* qrow = qp + (size_t)(b2 * H_ + h) * DH_;
  const float* krow = kh + ((size_t)(b1 * L_ + l) * H_ + h) * DH_;
  float acc = 0.f;
  for (int d = 0; d < DH_; ++d) acc += qrow[d] * krow[d];
  const float logit = acc * 0.125f;                // 1/sqrt(64)

  __shared__ float red[128];
  red[l] = logit;
  __syncthreads();
  for (int s = 64; s > 0; s >>= 1) {
    if (l < s) red[l] = fmaxf(red[l], red[l + s]);
    __syncthreads();
  }
  const float mx = red[0];
  __syncthreads();
  const float e = __expf(logit - mx);
  red[l] = e;
  __syncthreads();
  for (int s = 64; s > 0; s >>= 1) {
    if (l < s) red[l] += red[l + s];
    __syncthreads();
  }
  attn[(size_t)blk * L_ + l] = f2bf(e / red[0]);
}

// ---------------------------------------------------------------------------
// K7: context GEMM per (b1,h) via WMMA (pass 2 over kv)
//     M=16 (8 b2 + 8 pad), K=128, N=4096.
//     LDS tile stored COLUMN-major [col][k] with k-pairs packed per b32 store
//     -> B fragments are 2x ds_load_b128.
// ---------------------------------------------------------------------------
__global__ __launch_bounds__(256) void ctx_wmma_kernel(
    const float* __restrict__ kv, const unsigned short* __restrict__ attn,
    unsigned short* __restrict__ ctxcat) {
  const int bid  = blockIdx.x;
  const int nblk = bid & 15;
  const int bh   = bid >> 4;                       // b1*H + h
  const int h    = bh & (H_ - 1);
  const int b1   = bh >> 4;
  const int tid  = threadIdx.x;
  const int lane = tid & 31;
  const int wave = tid >> 5;
  const int n0   = nblk * 256;

  __shared__ unsigned short Bt[256 * 32];          // [col][k], 16 KB

  const unsigned short* arow_base = attn + (size_t)bh * B_ * L_;
  const int mrow  = lane & 15;
  const int khalf = (lane >= 16) ? 8 : 0;
  const int kb0   = (lane >= 16) ? 16 : 0;

  v8f c0 = (v8f)0.0f, c1 = (v8f)0.0f;

  for (int k0 = 0; k0 < L_; k0 += 32) {
    __syncthreads();
    // stage 32(k) x 256(col): pair-pack rows (2j, 2j+1) per column into b32
#pragma unroll
    for (int it = 0; it < 4; ++it) {
      const int u  = it * 256 + tid;               // 0..1023 pair-units
      const int j  = u >> 6;                       // k-pair 0..15
      const int c4 = (u & 63) * 4;                 // col 0..252
      const int ncol = n0 + c4;
      const int t = ncol >> 6;
      const int d = ncol & 63;
      const float* s0 = kv +
          (((size_t)(b1 * L_ + k0 + 2 * j) * T_ + t) * F_) + h * DH_ + d;
      const float* s1 = s0 + (size_t)T_ * F_;
      if (k0 + 32 < L_) __builtin_prefetch(s0 + (size_t)32 * T_ * F_, 0, 1);
      const float4 v0 = *(const float4*)s0;
      const float4 v1 = *(const float4*)s1;
      const unsigned p0 = (unsigned)f2bf(v0.x) | ((unsigned)f2bf(v1.x) << 16);
      const unsigned p1 = (unsigned)f2bf(v0.y) | ((unsigned)f2bf(v1.y) << 16);
      const unsigned p2 = (unsigned)f2bf(v0.z) | ((unsigned)f2bf(v1.z) << 16);
      const unsigned p3 = (unsigned)f2bf(v0.w) | ((unsigned)f2bf(v1.w) << 16);
      *(unsigned*)&Bt[(c4 + 0) * 32 + 2 * j] = p0;
      *(unsigned*)&Bt[(c4 + 1) * 32 + 2 * j] = p1;
      *(unsigned*)&Bt[(c4 + 2) * 32 + 2 * j] = p2;
      *(unsigned*)&Bt[(c4 + 3) * 32 + 2 * j] = p3;
    }
    __syncthreads();

    // A fragment (attention rows; pad rows >= 8 are zero)
    v16us au = (v16us)(unsigned short)0;
    if (mrow < 8) {
      au = load_a_frag(arow_base + (size_t)mrow * L_, k0, khalf);
    }
    const v16bf a = as_bf(au);

#pragma unroll
    for (int s = 0; s < 2; ++s) {
      const int c = wave * 32 + s * 16 + (lane & 15);
      const v16bf b = as_bf(load_b_frag(&Bt[c * 32 + kb0]));
      v8f& acc = s ? c1 : c0;
      acc = __builtin_amdgcn_wmma_f32_16x16x32_bf16(false, a, false, b,
                                                    (short)0, acc, false, false);
    }
  }

  // scatter-store into ctxcat[b1][t][b2][h*64+d] (bf16), skip pad rows
#pragma unroll
  for (int s = 0; s < 2; ++s) {
    const v8f& acc = s ? c1 : c0;
    const int ncol = n0 + wave * 32 + s * 16 + (lane & 15);
    const int t = ncol >> 6;
    const int d = ncol & 63;
#pragma unroll
    for (int r = 0; r < 8; ++r) {
      const int m = r + ((lane >= 16) ? 8 : 0);    // b2
      if (m < 8) {
        ctxcat[(((size_t)b1 * T_ + t) * B_ + m) * F_ + h * DH_ + d] =
            f2bf(acc[r]);
      }
    }
  }
}

// ---------------------------------------------------------------------------
// K8: out[m, i] = sum_k ctxcat[m, k] * W2[i, k] + bvec[i]
//     M=4096, K=1024, N=1024.
//     A tile (16 x 1024 bf16, 32 KB) is async-staged into LDS ONCE per block
//     (global_load_async_to_lds_b128) -> A fragments are 2x ds_load_b128,
//     removing the 8x redundant global A reads.  B (W2) fragments load
//     straight from L2-resident global memory as contiguous b128 pairs.
// ---------------------------------------------------------------------------
__global__ __launch_bounds__(256) void out_wmma_kernel(
    const unsigned short* __restrict__ A, const unsigned short* __restrict__ Bw,
    const float* __restrict__ bvec, float* __restrict__ out) {
  __shared__ unsigned short As[16 * F_];           // 32 KB
  const int bid   = blockIdx.x;
  const int nblk  = bid & 3;
  const int mtile = bid >> 2;
  const int lane  = threadIdx.x & 31;
  const int wave  = threadIdx.x >> 5;
  const int m0    = mtile * 16;
  const int n0    = nblk * 256;
  const int khalf = (lane >= 16) ? 8 : 0;
  const int kb0   = (lane >= 16) ? 16 : 0;

  {  // async-stage the A tile: 2048 x 16B pieces, 8 per thread
    const unsigned lds_base = (unsigned)(uintptr_t)(&As[0]);
    const unsigned long long gbase =
        (unsigned long long)(uintptr_t)(A + (size_t)m0 * F_);
#pragma unroll
    for (int it = 0; it < 8; ++it) {
      const unsigned byteoff = (unsigned)((it * 256 + threadIdx.x) * 16);
      asm volatile("global_load_async_to_lds_b128 %0, %1, %2"
                   :: "v"(lds_base + byteoff), "v"(byteoff), "s"(gbase)
                   : "memory");
    }
    asm volatile("s_wait_asynccnt 0x0" ::: "memory");
    __syncthreads();
  }

  const unsigned short* arow = As + (size_t)(lane & 15) * F_;
  const unsigned short* bcol0 =
      Bw + (size_t)(n0 + wave * 32 + (lane & 15)) * F_;
  const unsigned short* bcol1 = bcol0 + (size_t)16 * F_;

  v8f c0 = (v8f)0.0f, c1 = (v8f)0.0f;

  for (int k0 = 0; k0 < F_; k0 += 32) {
    const v16bf a = as_bf(load_a_frag(arow, k0, khalf));
    c0 = __builtin_amdgcn_wmma_f32_16x16x32_bf16(false, a, false,
             as_bf(load_b_frag(bcol0 + k0 + kb0)), (short)0, c0, false, false);
    c1 = __builtin_amdgcn_wmma_f32_16x16x32_bf16(false, a, false,
             as_bf(load_b_frag(bcol1 + k0 + kb0)), (short)0, c1, false, false);
  }

#pragma unroll
  for (int s = 0; s < 2; ++s) {
    const v8f& acc = s ? c1 : c0;
    const int ncol = n0 + wave * 32 + s * 16 + (lane & 15);
#pragma unroll
    for (int r = 0; r < 8; ++r) {
      const int m = m0 + r + ((lane >= 16) ? 8 : 0);
      out[(size_t)m * F_ + ncol] = acc[r] + bvec[ncol];
    }
  }
}

// ---------------------------------------------------------------------------
// Host-side launcher
// ---------------------------------------------------------------------------
extern "C" void kernel_launch(void* const* d_in, const int* in_sizes, int n_in,
                              void* d_out, int out_size, void* d_ws, size_t ws_size,
                              hipStream_t stream) {
  (void)in_sizes; (void)n_in; (void)out_size; (void)ws_size;

  const float* q  = (const float*)d_in[0];
  const float* kv = (const float*)d_in[1];
  const float* Wq = (const float*)d_in[2];
  const float* bq = (const float*)d_in[3];
  const float* Wk = (const float*)d_in[4];
  const float* bk = (const float*)d_in[5];
  const float* Wv = (const float*)d_in[6];
  const float* bv = (const float*)d_in[7];
  const float* Wo = (const float*)d_in[8];
  const float* bo = (const float*)d_in[9];
  float* out = (float*)d_out;

  char* ws = (char*)d_ws;
  size_t off = 0;
  auto take = [&](size_t bytes) -> char* {
    char* p = ws + off;
    off += (bytes + 255) & ~(size_t)255;
    return p;
  };
  unsigned short* km_bf  = (unsigned short*)take((size_t)B_ * L_ * F_ * 2);
  float*          qp     = (float*)take((size_t)B_ * H_ * DH_ * 4);
  unsigned short* WkBt   = (unsigned short*)take((size_t)DH_ * DH_ * 2);
  float*          kh     = (float*)take((size_t)B_ * L_ * H_ * DH_ * 4);
  unsigned short* attn   = (unsigned short*)take((size_t)B_ * H_ * B_ * L_ * 2);
  unsigned short* W2bf   = (unsigned short*)take((size_t)F_ * F_ * 2);
  float*          bvec   = (float*)take((size_t)F_ * 4);
  unsigned short* ctxcat = (unsigned short*)take((size_t)B_ * T_ * B_ * F_ * 2);

  mean_kv_kernel<<<B_ * L_, 256, 0, stream>>>(kv, km_bf);
  qproj_kernel<<<(B_ * H_ * DH_) / 256, 256, 0, stream>>>(q, Wq, bq, qp);
  wkpack_kernel<<<(DH_ * DH_) / 256, 256, 0, stream>>>(Wk, WkBt);
  wcomb_kernel<<<F_, 64, 0, stream>>>(Wv, bv, Wo, bo, W2bf, bvec);
  kproj_wmma_kernel<<<(B_ * L_ * H_) / 64, 128, 0, stream>>>(km_bf, WkBt, bk, kh);
  attn_softmax_kernel<<<B_ * H_ * B_, L_, 0, stream>>>(qp, kh, attn);
  ctx_wmma_kernel<<<B_ * H_ * 16, 256, 0, stream>>>(kv, attn, ctxcat);
  out_wmma_kernel<<<(B_ * T_ * B_ / 16) * 4, 256, 0, stream>>>(ctxcat, W2bf, bvec, out);
}
